// AttentionBlock_23768349016409
// MI455X (gfx1250) — matrix operand order repaired and checked by reading, not compiled, not run on hardware
//
#include <hip/hip_runtime.h>
#include <hip/hip_bf16.h>
#include <math.h>

// ---------------------------------------------------------------------------
// AttentionBlock, MI455X (gfx1250), wave32.
// Attention is degenerate (softmax over length-1 axis) -> o == v, so the block
// collapses to:  out = W_eff @ rmsnorm(x) + b_eff + x,  W_eff = W_proj @ W_v.
// Main GEMM [32768 x 512] x [512 x 512] done with v_wmma_f32_16x16x32_bf16
// using a bf16 hi/lo split (hi*hi + hi*lo + lo*hi) for ~fp32 accuracy.
// x tiles are DMA'd into LDS with the Tensor Data Mover (tensor_load_to_lds).
// ---------------------------------------------------------------------------

#define DIM    512
#define NPOS   32768            // T*H*W = 8*64*64
#define MTILE  32               // positions per workgroup
#define YPAD   520              // padded bf16 row length (bank-conflict relief)
#define GRID_MAIN (NPOS / MTILE)

typedef __attribute__((ext_vector_type(16))) __bf16   v16bf;
typedef __attribute__((ext_vector_type(8)))  float    v8f;
typedef __attribute__((ext_vector_type(4)))  unsigned v4u;
typedef __attribute__((ext_vector_type(8)))  int      v8i;
typedef __attribute__((ext_vector_type(4)))  int      v4i;

union Frag { v16bf bf; v4u u[2]; };

static __device__ __forceinline__ unsigned short f32_to_bf16_rne(float f) {
  unsigned u = __float_as_uint(f);
  u += 0x7FFFu + ((u >> 16) & 1u);
  return (unsigned short)(u >> 16);
}
static __device__ __forceinline__ float bf16_bits_to_f32(unsigned short h) {
  return __uint_as_float((unsigned)h << 16);
}
static __device__ __forceinline__ unsigned lds_byte_addr(void* p) {
  typedef __attribute__((address_space(3))) char lds_char;
  return (unsigned)(unsigned long long)(lds_char*)p;  // LDS byte offset
}

// ---------------------------------------------------------------------------
// Prep 1: W_eff[o][c] = sum_k w_proj[o][k] * w_qkv[1024+k][c]  (V part only),
// stored as bf16 hi/lo pair.  64 blocks x 512 threads, 8 output rows/block.
// ---------------------------------------------------------------------------
__global__ __launch_bounds__(512) void prep_weights(
    const float* __restrict__ w_qkv, const float* __restrict__ w_proj,
    unsigned short* __restrict__ Wh, unsigned short* __restrict__ Wl) {
  __shared__ float wp[8][DIM];           // 16 KB: w_proj rows for this block
  const int t  = threadIdx.x;
  const int bo = blockIdx.x * 8;
  for (int i = t; i < 8 * DIM; i += 512)
    wp[i >> 9][i & 511] = w_proj[(size_t)(bo + (i >> 9)) * DIM + (i & 511)];
  __syncthreads();

  float acc[8] = {0.f, 0.f, 0.f, 0.f, 0.f, 0.f, 0.f, 0.f};
  for (int k = 0; k < DIM; ++k) {
    const float wv = w_qkv[(size_t)(2 * DIM + k) * DIM + t];   // V rows
#pragma unroll
    for (int j = 0; j < 8; ++j) acc[j] += wp[j][k] * wv;
  }
#pragma unroll
  for (int j = 0; j < 8; ++j) {
    const unsigned short hi = f32_to_bf16_rne(acc[j]);
    const unsigned short lo = f32_to_bf16_rne(acc[j] - bf16_bits_to_f32(hi));
    Wh[(size_t)(bo + j) * DIM + t] = hi;
    Wl[(size_t)(bo + j) * DIM + t] = lo;
  }
}

// ---------------------------------------------------------------------------
// Prep 2: b_eff[o] = w_proj[o][:] . b_qkv[1024:1536] + b_proj[o]
// ---------------------------------------------------------------------------
__global__ __launch_bounds__(512) void prep_bias(
    const float* __restrict__ w_proj, const float* __restrict__ b_qkv,
    const float* __restrict__ b_proj, float* __restrict__ beff) {
  const int o = threadIdx.x;
  float b = b_proj[o];
  for (int k = 0; k < DIM; ++k) b += w_proj[(size_t)o * DIM + k] * b_qkv[2 * DIM + k];
  beff[o] = b;
}

// ---------------------------------------------------------------------------
// Main kernel: per workgroup, 32 positions x 512 channels.
// 8 waves: wave w -> mgrp = w&1 (16 rows), ngrp = w>>1 (128 output cols).
// ---------------------------------------------------------------------------
__global__ __launch_bounds__(256) void attn_main(
    const float* __restrict__ x, const float* __restrict__ gamma,
    const unsigned short* __restrict__ Wh, const unsigned short* __restrict__ Wl,
    const float* __restrict__ beff, float* __restrict__ out) {

  __shared__ struct {
    float Xf[DIM][MTILE];                                   // 64 KB: x tile (channel-major)
    union {
      struct { unsigned short Yh[MTILE][YPAD];
               unsigned short Yl[MTILE][YPAD]; } y;         // 65 KB: bf16 hi/lo activations
      float O[DIM][33];                                     // 66 KB: result transpose buffer
    } u;
    float part[8][MTILE];
    float rs[MTILE];
    float gam[DIM];
    float bs[DIM];
  } sm;

  const int tid  = threadIdx.x;
  const int lane = tid & 31;
  const int wid  = tid >> 5;

  // ---- Issue TDM: DMA x tile [512 rows x 32 floats], row stride 32768 floats.
  if (tid < 32) {
    const unsigned long long ga =
        (unsigned long long)(const void*)(x + (size_t)blockIdx.x * MTILE);
    const unsigned lds = lds_byte_addr(&sm.Xf[0][0]);
    v4u g0;
    g0[0] = 1u;                                             // count=1, user D#
    g0[1] = lds;                                            // lds_addr
    g0[2] = (unsigned)(ga & 0xFFFFFFFFull);                 // global_addr lo
    g0[3] = (unsigned)((ga >> 32) & 0x01FFFFFFull) | 0x80000000u; // addr hi | type=2
    v8i g1;
    g1[0] = (int)(2u << 16);                                // data_size = 4 bytes
    g1[1] = (int)(((unsigned)NPOS & 0xFFFFu) << 16);        // tensor_dim0[15:0]
    g1[2] = (int)((((unsigned)NPOS >> 16) & 0xFFFFu) | ((unsigned)DIM << 16)); // td0 hi | td1 lo
    g1[3] = (int)((unsigned)MTILE << 16);                   // td1 hi=0 | tile_dim0=32
    g1[4] = (int)DIM;                                       // tile_dim1=512 | tile_dim2=0
    g1[5] = (int)NPOS;                                      // tensor_dim0_stride lo
    g1[6] = 0;                                              // stride0 hi | stride1 lo
    g1[7] = 0;
    v4i g2; g2[0] = 0; g2[1] = 0; g2[2] = 0; g2[3] = 0;
    v4i g3 = g2;
    v8i g4; g4[0] = 0; g4[1] = 0; g4[2] = 0; g4[3] = 0;
            g4[4] = 0; g4[5] = 0; g4[6] = 0; g4[7] = 0;
    __builtin_amdgcn_tensor_load_to_lds(g0, g1, g2, g3, g4, 0);
  }

  // Stage gamma / fused bias while the DMA runs.
  sm.gam[tid]       = gamma[tid];
  sm.gam[tid + 256] = gamma[tid + 256];
  sm.bs[tid]        = beff[tid];
  sm.bs[tid + 256]  = beff[tid + 256];

  if (tid < 32) __builtin_amdgcn_s_wait_tensorcnt(0);
  __syncthreads();

  // ---- RMS norm factors: 8 partial sums of 64 channels per position.
  const int m32 = tid & 31;
  const int cg  = tid >> 5;
  float ss = 0.f;
  for (int c = cg * 64; c < cg * 64 + 64; ++c) { const float v = sm.Xf[c][m32]; ss += v * v; }
  sm.part[cg][m32] = ss;
  __syncthreads();
  if (tid < 32) {
    float t = 0.f;
#pragma unroll
    for (int j = 0; j < 8; ++j) t += sm.part[j][tid];
    const float nrm = fmaxf(sqrtf(t), 1e-12f);
    sm.rs[tid] = 22.627416997969522f / nrm;                 // sqrt(512)/clip(||x||)
  }
  __syncthreads();

  // ---- Build bf16 hi/lo activation tile, [m][k] row-major.
  for (int i = 0; i < 64; ++i) {
    const int c = i * 8 + cg;
    const float v = sm.Xf[c][m32] * sm.rs[m32] * sm.gam[c];
    const unsigned short hi = f32_to_bf16_rne(v);
    const unsigned short lo = f32_to_bf16_rne(v - bf16_bits_to_f32(hi));
    sm.u.y.Yh[m32][c] = hi;
    sm.u.y.Yl[m32][c] = lo;
  }
  __syncthreads();

  // ---- GEMM: D[m][n] = sum_k Y[m][k] * W_eff[n][k]; bf16x3 products.
  const int mgrp  = wid & 1;
  const int ngrp  = wid >> 1;
  const int nbase = ngrp * 128;
  const int mA    = (lane & 15) + mgrp * 16;
  const int koffA = (lane >> 4) * 8;      // A 16x32 bf16 layout: K runs {koffA..+7, +16..+23}
  const int nB    = lane & 15;
  const int koffB = (lane >> 4) * 8;      // B 32x16 bf16 layout: K runs {koffB..+7, +16..+23}

  const unsigned short* WhB = Wh + (size_t)((nbase + nB) * DIM + koffB);
  const unsigned short* WlB = Wl + (size_t)((nbase + nB) * DIM + koffB);

  const v8f zero8 = {0.f, 0.f, 0.f, 0.f, 0.f, 0.f, 0.f, 0.f};
  v8f acc[8];
#pragma unroll
  for (int nt = 0; nt < 8; ++nt) acc[nt] = zero8;

  for (int ks = 0; ks < 16; ++ks) {
    const int k0 = ks * 32;
    Frag ah, al;
    { const v4u* pa = (const v4u*)&sm.u.y.Yh[mA][k0 + koffA]; ah.u[0] = pa[0]; ah.u[1] = pa[2]; }
    { const v4u* pl = (const v4u*)&sm.u.y.Yl[mA][k0 + koffA]; al.u[0] = pl[0]; al.u[1] = pl[2]; }
#pragma unroll
    for (int nt = 0; nt < 8; ++nt) {
      const v4u* qh = (const v4u*)(WhB + k0 + nt * (16 * DIM));
      const v4u* ql = (const v4u*)(WlB + k0 + nt * (16 * DIM));
      Frag bh, bl;
      bh.u[0] = qh[0]; bh.u[1] = qh[2];
      bl.u[0] = ql[0]; bl.u[1] = ql[2];
      acc[nt] = __builtin_amdgcn_wmma_f32_16x16x32_bf16(false, ah.bf, false, bh.bf,
                                                        (short)0, acc[nt], false, false);
      acc[nt] = __builtin_amdgcn_wmma_f32_16x16x32_bf16(false, al.bf, false, bh.bf,
                                                        (short)0, acc[nt], false, false);
      acc[nt] = __builtin_amdgcn_wmma_f32_16x16x32_bf16(false, ah.bf, false, bl.bf,
                                                        (short)0, acc[nt], false, false);
    }
  }

  __syncthreads();   // Y no longer needed; O aliases it.

  // ---- Spill accumulators (+bias) to LDS, transposed to channel-major.
  const int khi = lane >> 4;
#pragma unroll
  for (int nt = 0; nt < 8; ++nt) {
    const int n  = nbase + nt * 16 + nB;
    const float b = sm.bs[n];
#pragma unroll
    for (int r = 0; r < 8; ++r) {
      const int m = mgrp * 16 + khi * 8 + r;   // D layout: VGPR r -> M = r + 8*(lane>>4)
      sm.u.O[n][m] = acc[nt][r] + b;
    }
  }
  __syncthreads();

  // ---- Residual add + coalesced store (128B per channel row).
  const size_t p0 = (size_t)blockIdx.x * MTILE;
  for (int i = 0; i < 64; ++i) {
    const int idx = i * 256 + tid;
    const int c = idx >> 5;
    const int m = idx & 31;
    out[(size_t)c * NPOS + p0 + m] = sm.u.O[c][m] + sm.Xf[c][m];
  }
}

// ---------------------------------------------------------------------------
extern "C" void kernel_launch(void* const* d_in, const int* in_sizes, int n_in,
                              void* d_out, int out_size, void* d_ws, size_t ws_size,
                              hipStream_t stream) {
  (void)in_sizes; (void)n_in; (void)out_size; (void)ws_size;
  const float* x      = (const float*)d_in[0];
  const float* gamma  = (const float*)d_in[1];
  const float* w_qkv  = (const float*)d_in[2];
  const float* b_qkv  = (const float*)d_in[3];
  const float* w_proj = (const float*)d_in[4];
  const float* b_proj = (const float*)d_in[5];
  float* out = (float*)d_out;

  char* ws = (char*)d_ws;
  unsigned short* Wh = (unsigned short*)ws;                        // 512 KB bf16 hi
  unsigned short* Wl = (unsigned short*)(ws + (size_t)DIM * DIM * 2); // 512 KB bf16 lo
  float*          be = (float*)(ws + (size_t)DIM * DIM * 4);       // 2 KB fused bias

  prep_weights<<<DIM / 8, 512, 0, stream>>>(w_qkv, w_proj, Wh, Wl);
  prep_bias<<<1, 512, 0, stream>>>(w_proj, b_qkv, b_proj, be);
  attn_main<<<GRID_MAIN, 256, 0, stream>>>(x, gamma, Wh, Wl, be, out);
}
